// Attention_81028853006805
// MI455X (gfx1250) — compile-verified, matrix operand order checked by reading, not run on hardware
//
#include <hip/hip_runtime.h>

// ---------------------------------------------------------------------------
// Multi-head attention block for MI455X (gfx1250), bf16 WMMA + f32 accumulate.
// x[8,1024,768] -> qkv gemm -> flash attention (12 heads, hd=64) -> proj gemm.
// Entire bf16 working set (~68 MB) is L2-resident (192 MB L2), so GEMM
// fragments are loaded directly from global memory with B128 loads.
// GEMM mainloops use ping-pong register double buffering (unroll-by-2, no
// buffer-rotation movs) on 32x64 wave tiles: 6 fragment loads per 8 WMMAs.
// ---------------------------------------------------------------------------

typedef __bf16 bf;
typedef __attribute__((ext_vector_type(16))) __bf16 v16bf;
typedef __attribute__((ext_vector_type(8)))  __bf16 v8bf;
typedef __attribute__((ext_vector_type(4)))  __bf16 v4bf;
typedef __attribute__((ext_vector_type(8)))  float  v8f;

#define DIMC   768
#define SEQ    1024
#define BATCH  8
#define NHEADS 12
#define HD     64
#define MROWS  (BATCH * SEQ)      // 8192
#define NQKV   (3 * DIMC)         // 2304

// softmax uses exp2; fold SCALE (=hd^-0.5=0.125) * log2(e) into Q at QKV epilogue
#define QSCALE (0.125f * 1.4426950408889634f)

__device__ __forceinline__ v8f zero8() {
  v8f z = {0.f, 0.f, 0.f, 0.f, 0.f, 0.f, 0.f, 0.f};
  return z;
}

// Load a 16x32 bf16 WMMA A/B fragment per the CDNA5 ISA VGPR layout:
//   lanes 0-15 / 16-31 both cover rows 0-15 (row = lane&15);
//   VGPRs 0-3 hold K = half*8 .. half*8+7, VGPRs 4-7 hold K = 16+half*8 .. +7.
// 'base' points at (row0, k0); 'stride' = row stride in elements.
__device__ __forceinline__ v16bf load_frag(const bf* __restrict__ base,
                                           int stride, int lane) {
  const int r = lane & 15;
  const int h = (lane >> 4) & 1;
  const bf* p = base + (size_t)r * stride + h * 8;
  union { v16bf f; v8bf v[2]; } u;
  u.v[0] = *reinterpret_cast<const v8bf*>(p);       // global_load_b128
  u.v[1] = *reinterpret_cast<const v8bf*>(p + 16);  // global_load_b128
  return u.f;
}

__device__ __forceinline__ v8f wmma_bf16(v16bf a, v16bf b, v8f c) {
  return __builtin_amdgcn_wmma_f32_16x16x32_bf16(
      /*neg_a=*/false, a, /*neg_b=*/false, b,
      /*c_mod=*/(short)0, c, /*reuse_a=*/false, /*reuse_b=*/false);
}

// reductions across the 16 lanes that share one C-matrix row (lane bits 0-3)
__device__ __forceinline__ float rowmax16(float v) {
  v = fmaxf(v, __shfl_xor(v, 1, 32));
  v = fmaxf(v, __shfl_xor(v, 2, 32));
  v = fmaxf(v, __shfl_xor(v, 4, 32));
  v = fmaxf(v, __shfl_xor(v, 8, 32));
  return v;
}
__device__ __forceinline__ float rowsum16(float v) {
  v += __shfl_xor(v, 1, 32);
  v += __shfl_xor(v, 2, 32);
  v += __shfl_xor(v, 4, 32);
  v += __shfl_xor(v, 8, 32);
  return v;
}

// ---------------------------------------------------------------------------
// 32x64 GEMM mainloop helpers: acc[0..3] = rows m0..m0+15 x cols t*16,
// acc[4..7] = rows m0+16..m0+31.
// ---------------------------------------------------------------------------
__device__ __forceinline__ void load_set(const bf* __restrict__ arow0,
                                         const bf* __restrict__ arow1,
                                         const bf* __restrict__ wbase, int k,
                                         int lane, v16bf& a0, v16bf& a1,
                                         v16bf b[4]) {
  a0 = load_frag(arow0 + k, DIMC, lane);
  a1 = load_frag(arow1 + k, DIMC, lane);
#pragma unroll
  for (int t = 0; t < 4; ++t)
    b[t] = load_frag(wbase + (size_t)(t * 16) * DIMC + k, DIMC, lane);
}

__device__ __forceinline__ void wmma_set(v16bf a0, v16bf a1, const v16bf b[4],
                                         v8f acc[8]) {
#pragma unroll
  for (int t = 0; t < 4; ++t) acc[t] = wmma_bf16(a0, b[t], acc[t]);
#pragma unroll
  for (int t = 0; t < 4; ++t) acc[4 + t] = wmma_bf16(a1, b[t], acc[4 + t]);
}

// Ping-pong double buffered over K (two disjoint register sets, no rotation
// movs). DIMC/32 = 24 K-steps: 11 loop iterations of 2 steps + peeled tail.
__device__ __forceinline__ void gemm_tile_32x64(const bf* __restrict__ arow0,
                                                const bf* __restrict__ arow1,
                                                const bf* __restrict__ wbase,
                                                int lane, v8f acc[8]) {
  v16bf a0A, a1A, bA[4];
  v16bf a0B, a1B, bB[4];
  load_set(arow0, arow1, wbase, 0, lane, a0A, a1A, bA);
#pragma unroll 1
  for (int k = 0; k < DIMC - 64; k += 64) {
    load_set(arow0, arow1, wbase, k + 32, lane, a0B, a1B, bB);
    wmma_set(a0A, a1A, bA, acc);
    load_set(arow0, arow1, wbase, k + 64, lane, a0A, a1A, bA);
    wmma_set(a0B, a1B, bB, acc);
  }
  load_set(arow0, arow1, wbase, DIMC - 32, lane, a0B, a1B, bB);
  wmma_set(a0A, a1A, bA, acc);
  wmma_set(a0B, a1B, bB, acc);
}

// ---------------------------------------------------------------------------
// f32 -> bf16 conversion (float4 vectorized, all sizes are multiples of 4)
// ---------------------------------------------------------------------------
__global__ void __launch_bounds__(256)
cvt_f32_bf16(const float* __restrict__ src, bf* __restrict__ dst, int n) {
  int i = (blockIdx.x * blockDim.x + threadIdx.x) * 4;
  if (i + 3 < n) {
    float4 v = *reinterpret_cast<const float4*>(src + i);
    v4bf o = {(bf)v.x, (bf)v.y, (bf)v.z, (bf)v.w};
    *reinterpret_cast<v4bf*>(dst + i) = o;
  }
}

// ---------------------------------------------------------------------------
// QKV GEMM: [8192,768] @ [2304,768]^T + bias. Per-wave 32x64 output tile.
// Writes Q (pre-scaled by SCALE*log2e), K row-major [B,H,P,Hd] and V
// transposed [B,H,Hd,P], all bf16.
// ---------------------------------------------------------------------------
__global__ void __launch_bounds__(256)
qkv_gemm(const bf* __restrict__ xb,     // [8192, 768]
         const bf* __restrict__ wb,     // [2304, 768]
         const float* __restrict__ bias,// [2304]
         bf* __restrict__ qb,           // [B,H,P,Hd]
         bf* __restrict__ kb,           // [B,H,P,Hd]
         bf* __restrict__ vtb) {        // [B,H,Hd,P]
  const int lane = threadIdx.x & 31;
  const int wave = threadIdx.x >> 5;
  const int tile = blockIdx.x * 8 + wave;
  const int NT = NQKV / 64;                   // 36 n-tiles of width 64
  if (tile >= (MROWS / 32) * NT) return;      // wave-uniform
  const int m0 = (tile / NT) * 32;
  const int n0 = (tile % NT) * 64;

  v8f acc[8] = {zero8(), zero8(), zero8(), zero8(),
                zero8(), zero8(), zero8(), zero8()};
  gemm_tile_32x64(xb + (size_t)m0 * DIMC, xb + (size_t)(m0 + 16) * DIMC,
                  wb + (size_t)n0 * DIMC, lane, acc);

  // Epilogue: C layout -> lane holds col n = lane&15, rows r + 8*(lane>>4).
  const int nl = lane & 15;
  const int half = lane >> 4;
  const int b = m0 >> 10;       // batch (32-row tile never crosses batch)
  const int p0 = m0 & 1023;     // seq position base
#pragma unroll
  for (int t = 0; t < 4; ++t) {
    const int n = n0 + t * 16 + nl;
    const int which = n / DIMC;            // 0=q 1=k 2=v (uniform per 64-tile)
    const int rem = n - which * DIMC;
    const int hh = rem >> 6;
    const int hd = rem & 63;
    const float bv = bias[n];
    const size_t bh = (size_t)b * NHEADS + hh;
#pragma unroll
    for (int g = 0; g < 2; ++g) {          // two 16-row halves of the tile
#pragma unroll
      for (int r = 0; r < 8; ++r) {
        const int p = p0 + g * 16 + r + 8 * half;
        const float val = acc[g * 4 + t][r] + bv;
        if (which == 0) {
          qb[(bh * SEQ + p) * HD + hd] = (bf)(val * QSCALE);
        } else if (which == 1) {
          kb[(bh * SEQ + p) * HD + hd] = (bf)val;
        } else {
          vtb[(bh * HD + hd) * SEQ + p] = (bf)val;
        }
      }
    }
  }
}

// ---------------------------------------------------------------------------
// Flash attention: one wave per 16-row Q tile of one (batch, head).
// KV processed in chunks of 64. Base-2 online softmax. P re-laid out
// through a private per-wave LDS tile (C-frag layout -> A-frag layout).
// V fragments are loaded at the top of the chunk so their latency hides
// under the S WMMAs + softmax VALU work.
// ---------------------------------------------------------------------------
__global__ void __launch_bounds__(128)
attn_kernel(const bf* __restrict__ qb,   // [B,H,P,Hd]
            const bf* __restrict__ kb,   // [B,H,P,Hd]
            const bf* __restrict__ vtb,  // [B,H,Hd,P]
            bf* __restrict__ ob) {       // [B,P,H*Hd]
  __shared__ bf pls[4][16 * 72];         // per-wave 16x64 P tile, padded stride
  const int lane = threadIdx.x & 31;
  const int wave = threadIdx.x >> 5;
  const int gtile = blockIdx.x * 4 + wave;
  if (gtile >= BATCH * NHEADS * (SEQ / 16)) return;  // wave-uniform
  const int qt = gtile & 63;
  const int bh = gtile >> 6;             // b*NHEADS + h
  const int q0 = qt * 16;

  const bf* Q  = qb  + (size_t)bh * SEQ * HD;
  const bf* K  = kb  + (size_t)bh * SEQ * HD;
  const bf* Vt = vtb + (size_t)bh * HD * SEQ;

  // Q fragments for the whole tile (hd = 0..63 in two K-steps of 32)
  const v16bf aq0 = load_frag(Q + (size_t)q0 * HD +  0, HD, lane);
  const v16bf aq1 = load_frag(Q + (size_t)q0 * HD + 32, HD, lane);

  v8f o[4] = {zero8(), zero8(), zero8(), zero8()};
  float mst[8], lst[8];
#pragma unroll
  for (int r = 0; r < 8; ++r) { mst[r] = -1e30f; lst[r] = 0.f; }

  const int nl = lane & 15;
  const int half = lane >> 4;
  bf* pt = &pls[wave][0];

#pragma unroll 1
  for (int kv0 = 0; kv0 < SEQ; kv0 += 64) {
    // V fragments for this chunk: independent of S/softmax, issue first
    v16bf vf[2][4];
#pragma unroll
    for (int kk = 0; kk < 2; ++kk)
#pragma unroll
      for (int j = 0; j < 4; ++j)
        vf[kk][j] = load_frag(Vt + (size_t)(j * 16) * SEQ + kv0 + kk * 32,
                              SEQ, lane);

    // S = Q @ K^T  (already scaled: Q carries SCALE*log2e)
    v8f s[4] = {zero8(), zero8(), zero8(), zero8()};
#pragma unroll
    for (int t = 0; t < 4; ++t) {
      const bf* Kt = K + (size_t)(kv0 + t * 16) * HD;
      s[t] = wmma_bf16(aq0, load_frag(Kt, HD, lane), s[t]);
      s[t] = wmma_bf16(aq1, load_frag(Kt + 32, HD, lane), s[t]);
    }
    // online softmax (base 2); row = r + 8*half, cols striped over lanes 0-15
#pragma unroll
    for (int r = 0; r < 8; ++r) {
      float mx = fmaxf(fmaxf(s[0][r], s[1][r]), fmaxf(s[2][r], s[3][r]));
      mx = rowmax16(mx);
      const float mnew = fmaxf(mst[r], mx);
      const float alpha = __builtin_amdgcn_exp2f(mst[r] - mnew);
      float rs = 0.f;
#pragma unroll
      for (int t = 0; t < 4; ++t) {
        const float p = __builtin_amdgcn_exp2f(s[t][r] - mnew);
        s[t][r] = p;
        rs += p;
      }
      rs = rowsum16(rs);
      lst[r] = lst[r] * alpha + rs;
      mst[r] = mnew;
#pragma unroll
      for (int j = 0; j < 4; ++j) o[j][r] *= alpha;
    }
    // stage P into LDS (bf16), converting C-frag layout to row-major
#pragma unroll
    for (int r = 0; r < 8; ++r)
#pragma unroll
      for (int t = 0; t < 4; ++t)
        pt[(r + 8 * half) * 72 + t * 16 + nl] = (bf)s[t][r];
    __asm__ volatile("s_wait_dscnt 0" ::: "memory");  // intra-wave LDS RAW

    // O += P @ V : A frag from LDS, B frags already in registers
#pragma unroll
    for (int kk = 0; kk < 2; ++kk) {
      const bf* pa = pt + nl * 72 + kk * 32 + half * 8;
      union { v16bf f; v8bf v[2]; } ua;
      ua.v[0] = *reinterpret_cast<const v8bf*>(pa);       // ds_load_b128
      ua.v[1] = *reinterpret_cast<const v8bf*>(pa + 16);  // ds_load_b128
#pragma unroll
      for (int j = 0; j < 4; ++j) o[j] = wmma_bf16(ua.f, vf[kk][j], o[j]);
    }
    __asm__ volatile("" ::: "memory");  // keep LDS reads before next overwrite
  }

  // normalize and store to [B, P, H*Hd]
  const int bidx = bh / NHEADS;
  const int hh = bh - bidx * NHEADS;
  float inv[8];
#pragma unroll
  for (int r = 0; r < 8; ++r) inv[r] = 1.0f / lst[r];
#pragma unroll
  for (int j = 0; j < 4; ++j) {
    const int hd = j * 16 + nl;
#pragma unroll
    for (int r = 0; r < 8; ++r) {
      const int p = q0 + r + 8 * half;
      ob[((size_t)bidx * SEQ + p) * DIMC + hh * HD + hd] =
          (bf)(o[j][r] * inv[r]);
    }
  }
}

// ---------------------------------------------------------------------------
// Projection GEMM: [8192,768]bf16 @ [768,768]^T + bias -> f32 out
// ---------------------------------------------------------------------------
__global__ void __launch_bounds__(256)
proj_gemm(const bf* __restrict__ ab,      // [8192, 768]
          const bf* __restrict__ wb,      // [768, 768]
          const float* __restrict__ bias, // [768]
          float* __restrict__ out) {      // [8192, 768]
  const int lane = threadIdx.x & 31;
  const int wave = threadIdx.x >> 5;
  const int tile = blockIdx.x * 8 + wave;
  const int NT = DIMC / 64;                   // 12 n-tiles
  if (tile >= (MROWS / 32) * NT) return;      // wave-uniform
  const int m0 = (tile / NT) * 32;
  const int n0 = (tile % NT) * 64;

  v8f acc[8] = {zero8(), zero8(), zero8(), zero8(),
                zero8(), zero8(), zero8(), zero8()};
  gemm_tile_32x64(ab + (size_t)m0 * DIMC, ab + (size_t)(m0 + 16) * DIMC,
                  wb + (size_t)n0 * DIMC, lane, acc);

  const int nl = lane & 15;
  const int half = lane >> 4;
#pragma unroll
  for (int t = 0; t < 4; ++t) {
    const int n = n0 + t * 16 + nl;
    const float bv = bias[n];
#pragma unroll
    for (int g = 0; g < 2; ++g) {
#pragma unroll
      for (int r = 0; r < 8; ++r) {
        const int m = m0 + g * 16 + r + 8 * half;
        out[(size_t)m * DIMC + n] = acc[g * 4 + t][r] + bv;
      }
    }
  }
}

// ---------------------------------------------------------------------------
// launch
// ---------------------------------------------------------------------------
extern "C" void kernel_launch(void* const* d_in, const int* in_sizes, int n_in,
                              void* d_out, int out_size, void* d_ws,
                              size_t ws_size, hipStream_t stream) {
  (void)in_sizes; (void)n_in; (void)out_size; (void)ws_size;
  const float* x      = (const float*)d_in[0];  // [8,1024,768]
  const float* w_qkv  = (const float*)d_in[1];  // [2304,768]
  const float* b_qkv  = (const float*)d_in[2];  // [2304]
  const float* w_proj = (const float*)d_in[3];  // [768,768]
  const float* b_proj = (const float*)d_in[4];  // [768]
  float* out = (float*)d_out;

  // workspace layout (bytes), all offsets 256-aligned
  constexpr size_t SZ_XB  = (size_t)MROWS * DIMC * 2;   // 12,582,912
  constexpr size_t SZ_WQ  = (size_t)NQKV * DIMC * 2;    //  3,538,944
  constexpr size_t SZ_WP  = (size_t)DIMC * DIMC * 2;    //  1,179,648
  constexpr size_t SZ_HQ  = (size_t)BATCH * NHEADS * SEQ * HD * 2;
  char* ws = (char*)d_ws;
  bf* xb   = (bf*)(ws);
  bf* wqb  = (bf*)(ws + SZ_XB);
  bf* wpb  = (bf*)(ws + SZ_XB + SZ_WQ);
  bf* qb   = (bf*)(ws + SZ_XB + SZ_WQ + SZ_WP);
  bf* kb   = (bf*)(ws + SZ_XB + SZ_WQ + SZ_WP + SZ_HQ);
  bf* vtb  = (bf*)(ws + SZ_XB + SZ_WQ + SZ_WP + 2 * SZ_HQ);
  bf* aob  = (bf*)(ws + SZ_XB + SZ_WQ + SZ_WP + 3 * SZ_HQ);

  cvt_f32_bf16<<<(MROWS * DIMC) / 1024, 256, 0, stream>>>(x, xb, MROWS * DIMC);
  cvt_f32_bf16<<<(NQKV * DIMC) / 1024, 256, 0, stream>>>(w_qkv, wqb, NQKV * DIMC);
  cvt_f32_bf16<<<(DIMC * DIMC) / 1024, 256, 0, stream>>>(w_proj, wpb, DIMC * DIMC);

  // 256 m-tiles * 36 n-tiles = 9216 wave tiles / 8 waves = 1152 blocks
  qkv_gemm<<<1152, 256, 0, stream>>>(xb, wqb, b_qkv, qb, kb, vtb);

  // 8*12*64 = 6144 wave tiles / 4 waves = 1536 blocks
  attn_kernel<<<1536, 128, 0, stream>>>(qb, kb, vtb, aob);

  // 256 * 12 = 3072 wave tiles / 8 waves = 384 blocks
  proj_gemm<<<384, 256, 0, stream>>>(aob, wpb, b_proj, out);
}
